// GraphConvolution_90915867721729
// MI455X (gfx1250) — matrix operand order repaired and checked by reading, not compiled, not run on hardware
//
#include <hip/hip_runtime.h>
#include <hip/hip_bf16.h>

// ---------------------------------------------------------------------------
// GCN layer for MI455X (gfx1250, wave32, WMMA + TDM):
//   support = x @ W            (bf16 hi/lo split WMMA, f32 accumulate;
//                               A tiles streamed HBM->LDS by Tensor Data Mover)
//   out     = segsum_rows(vals * support[cols])   (L2-resident FP32 atomics)
//   out     = relu(out + b)
// ---------------------------------------------------------------------------

typedef __attribute__((ext_vector_type(16))) __bf16       v16bf;
typedef __attribute__((ext_vector_type(8)))  float        v8f;
typedef __attribute__((ext_vector_type(4)))  unsigned int u32x4;
typedef __attribute__((ext_vector_type(8)))  int          i32x8;
typedef __attribute__((ext_vector_type(4)))  int          i32x4;

#define GCN_N_NODES   100000
#define GCN_IN_DIM    256
#define GCN_OUT_DIM   128
#define GCN_ROW_TILES (GCN_N_NODES / 16)   // 6250, exact

#if __has_builtin(__builtin_amdgcn_tensor_load_to_lds) && \
    __has_builtin(__builtin_amdgcn_s_wait_tensorcnt)
#define GCN_USE_TDM 1
#else
#define GCN_USE_TDM 0
#endif

#if GCN_USE_TDM
// Issue one TDM load of a 16-row x 32-col f32 tile of x into LDS.
// D# per CDNA5 ISA ch.8:
//  group0: [count=1] [lds_addr] [global_addr 57b | type=2]
//  group1: data_size=2(4B), tensor_dim0=256, tensor_dim1=100000,
//          tile_dim0=32, tile_dim1=16, tensor_dim0_stride=256
__device__ __forceinline__ void gcn_tdm_load_tile(unsigned lds_off,
                                                  unsigned long long ga) {
  u32x4 g0 = {1u, lds_off, (unsigned)ga,
              ((unsigned)(ga >> 32) & 0x01FFFFFFu) | 0x80000000u};
  i32x8 g1 = {0x00020000,        // data_size = 2 (4-byte elements)
              0x01000000,        // tensor_dim0[15:0]=256 in bits [63:48]
              (int)0x86A00000,   // tensor_dim1[15:0]=0x86A0 in bits [95:80]
              0x00200001,        // tensor_dim1[31:16]=1, tile_dim0=32
              16,                // tile_dim1 = 16
              256,               // tensor_dim0_stride = 256
              0, 0};
  i32x4 z4 = {0, 0, 0, 0};
  i32x8 z8 = {0, 0, 0, 0, 0, 0, 0, 0};
  __builtin_amdgcn_tensor_load_to_lds(g0, g1, z4, z4, z8, 0);
}
#endif

// -------------------- Kernel 1: x @ W via bf16 WMMA ------------------------
// One wave computes a 16-row x 128-col strip of `support`.
// A (x rows): TDM-streamed into a per-wave double buffer in LDS (f32), then
//             converted to bf16 hi + lo residual in registers.
// B (W):      staged transposed in LDS as bf16 hi + lo once per workgroup.
// 3 WMMAs per (ktile, ntile): Ahi*Bhi + Alo*Bhi + Ahi*Blo (~near-f32 accuracy)
__global__ __launch_bounds__(256)
__attribute__((amdgpu_waves_per_eu(4)))
void gcn_gemm_wmma(
    const float* __restrict__ x, const float* __restrict__ W,
    float* __restrict__ support)
{
  __shared__ __attribute__((aligned(32))) __bf16 sWhi[GCN_OUT_DIM * GCN_IN_DIM];
  __shared__ __attribute__((aligned(32))) __bf16 sWlo[GCN_OUT_DIM * GCN_IN_DIM];
#if GCN_USE_TDM
  __shared__ __attribute__((aligned(16))) float sA[8][2][16 * 32];  // 32 KB
#endif

  // Stage W[k][n] -> sW{hi,lo}[n][k] (transposed, bf16 split)
  for (int idx = threadIdx.x; idx < GCN_IN_DIM * GCN_OUT_DIM; idx += 256) {
    const int k = idx >> 7;        // / OUT_DIM
    const int n = idx & 127;       // % OUT_DIM
    const float f = W[idx];
    const __bf16 h = (__bf16)f;
    sWhi[n * GCN_IN_DIM + k] = h;
    sWlo[n * GCN_IN_DIM + k] = (__bf16)(f - (float)h);
  }
  __syncthreads();

  const int wave = threadIdx.x >> 5;
  const int lane = threadIdx.x & 31;
  const int row_tile = blockIdx.x * 8 + wave;
  if (row_tile >= GCN_ROW_TILES) return;   // whole-wave guard: EXEC all-1s inside

  const int half = lane >> 4;              // 0: lanes 0-15, 1: lanes 16-31
  const int m    = row_tile * 16 + (lane & 15);

#if GCN_USE_TDM
  // Wave-uniform (SGPR) base addresses for the TDM descriptors.
  const unsigned long long xbase =
      (unsigned long long)(uintptr_t)(x + (size_t)(row_tile * 16) * GCN_IN_DIM);
  const unsigned ga_lo0 = __builtin_amdgcn_readfirstlane((unsigned)xbase);
  const unsigned ga_hi0 = __builtin_amdgcn_readfirstlane((unsigned)(xbase >> 32));
  const unsigned lds0 = __builtin_amdgcn_readfirstlane(
      (unsigned)(uintptr_t)&sA[wave][0][0]);
  const unsigned lds1 = __builtin_amdgcn_readfirstlane(
      (unsigned)(uintptr_t)&sA[wave][1][0]);
  const unsigned long long gabase =
      ((unsigned long long)ga_hi0 << 32) | ga_lo0;
  // Prologue: start streaming k-tile 0.
  gcn_tdm_load_tile(lds0, gabase);
#else
  const float* xrow = x + (size_t)m * GCN_IN_DIM;
#endif

  const v8f vzero = {};
  v8f acc[8];
#pragma unroll
  for (int nt = 0; nt < 8; ++nt) acc[nt] = vzero;

  // unroll 2 == one TDM double-buffer period; keeps the scheduling window
  // small enough to stay under 256 VGPRs with no scratch spills.
#pragma unroll 2
  for (int kt = 0; kt < 8; ++kt) {
    const int kbase = kt * 32;
    // A k order used here (contiguous, not AMD's table order): lane half selects K runs
    //   half=0: K = [0..15];  half=1: K = [16..31]
#if GCN_USE_TDM
    if (kt < 7) {  // stream next k-tile while computing this one
      gcn_tdm_load_tile((kt + 1) & 1 ? lds1 : lds0,
                        gabase + (unsigned long long)((kbase + 32) * 4));
      __builtin_amdgcn_s_wait_tensorcnt(1);  // k-tile kt is resident
    } else {
      __builtin_amdgcn_s_wait_tensorcnt(0);  // last tile: drain
    }
    const float* arow = &sA[wave][kt & 1][(lane & 15) * 32 + 16 * half];
    const float4 a0 = *(const float4*)(arow);
    const float4 a1 = *(const float4*)(arow + 4);
    const float4 a2 = *(const float4*)(arow + 8);
    const float4 a3 = *(const float4*)(arow + 12);
#else
    const int r1 = kbase + 16 * half;
    const float4 a0 = *(const float4*)(xrow + r1);
    const float4 a1 = *(const float4*)(xrow + r1 + 4);
    const float4 a2 = *(const float4*)(xrow + r1 + 8);
    const float4 a3 = *(const float4*)(xrow + r1 + 12);
#endif
    float av[16] = {a0.x, a0.y, a0.z, a0.w, a1.x, a1.y, a1.z, a1.w,
                    a2.x, a2.y, a2.z, a2.w, a3.x, a3.y, a3.z, a3.w};
    v16bf Ahi, Alo;
#pragma unroll
    for (int e = 0; e < 16; ++e) {
      const __bf16 h = (__bf16)av[e];
      Ahi[e] = h;
      Alo[e] = (__bf16)(av[e] - (float)h);
    }

    // B k order used here (contiguous, same as A): lanes 0-15 hold K=0..15, lanes 16-31 K=16..31
    const int kb2 = kbase + 16 * half;
#pragma unroll
    for (int nt = 0; nt < 8; ++nt) {
      const int n = nt * 16 + (lane & 15);
      const v16bf Bhi = *(const v16bf*)(sWhi + n * GCN_IN_DIM + kb2);
      const v16bf Blo = *(const v16bf*)(sWlo + n * GCN_IN_DIM + kb2);
      acc[nt] = __builtin_amdgcn_wmma_f32_16x16x32_bf16(
          false, Ahi, false, Bhi, (short)0, acc[nt], false, false);
      acc[nt] = __builtin_amdgcn_wmma_f32_16x16x32_bf16(
          false, Alo, false, Bhi, (short)0, acc[nt], false, false);
      acc[nt] = __builtin_amdgcn_wmma_f32_16x16x32_bf16(
          false, Ahi, false, Blo, (short)0, acc[nt], false, false);
    }
  }

  // f32 C/D layout: lane l, VGPR v -> M = v + 8*(l/16), N = l%16
#pragma unroll
  for (int nt = 0; nt < 8; ++nt) {
    const int col = nt * 16 + (lane & 15);
#pragma unroll
    for (int v = 0; v < 8; ++v) {
      const int mr = row_tile * 16 + v + 8 * half;
      support[(size_t)mr * GCN_OUT_DIM + col] = acc[nt][v];
    }
  }
  (void)m;
}

// -------------------- Kernel 0: zero the accumulator -----------------------
__global__ __launch_bounds__(256) void gcn_zero(float* __restrict__ out, int n4)
{
  const int i = blockIdx.x * blockDim.x + threadIdx.x;
  if (i >= n4) return;
  ((float4*)out)[i] = make_float4(0.f, 0.f, 0.f, 0.f);
}

// -------------------- Kernel 2: edge-wise SpMM (atomics) -------------------
// One wave per edge iteration: 32 lanes x float4 = one 128-wide row.
// support + out both fit in the 192 MB L2, so gathers & atomics run at L2 BW.
// Prefetch the next edge's row to hide gather latency behind atomics.
__global__ __launch_bounds__(256) void gcn_edge_spmm(
    const float* __restrict__ support, const int* __restrict__ rows,
    const int* __restrict__ cols, const float* __restrict__ vals,
    float* __restrict__ out, int n_edges)
{
  const int lane   = threadIdx.x & 31;
  const int wid    = (blockIdx.x * blockDim.x + threadIdx.x) >> 5;
  const int nwaves = (gridDim.x * blockDim.x) >> 5;
  for (int e = wid; e < n_edges; e += nwaves) {
    const int en = e + nwaves;
    if (en < n_edges) {
      const int cn = cols[en];
      __builtin_prefetch(support + (size_t)cn * GCN_OUT_DIM + lane * 4, 0, 1);
    }
    const int   r = rows[e];
    const int   c = cols[e];
    const float v = vals[e];
    const float4 g = ((const float4*)(support + (size_t)c * GCN_OUT_DIM))[lane];
    float* dst = out + (size_t)r * GCN_OUT_DIM + lane * 4;
    __hip_atomic_fetch_add(dst + 0, g.x * v, __ATOMIC_RELAXED, __HIP_MEMORY_SCOPE_AGENT);
    __hip_atomic_fetch_add(dst + 1, g.y * v, __ATOMIC_RELAXED, __HIP_MEMORY_SCOPE_AGENT);
    __hip_atomic_fetch_add(dst + 2, g.z * v, __ATOMIC_RELAXED, __HIP_MEMORY_SCOPE_AGENT);
    __hip_atomic_fetch_add(dst + 3, g.w * v, __ATOMIC_RELAXED, __HIP_MEMORY_SCOPE_AGENT);
  }
}

// -------------------- Kernel 3: bias + ReLU epilogue -----------------------
__global__ __launch_bounds__(256) void gcn_bias_relu(
    float* __restrict__ out, const float* __restrict__ b, int n4)
{
  const int i = blockIdx.x * blockDim.x + threadIdx.x;
  if (i >= n4) return;
  const int col = (i & 31) * 4;   // 32 float4 per 128-wide row
  float4 v = ((float4*)out)[i];
  v.x = fmaxf(v.x + b[col + 0], 0.f);
  v.y = fmaxf(v.y + b[col + 1], 0.f);
  v.z = fmaxf(v.z + b[col + 2], 0.f);
  v.w = fmaxf(v.w + b[col + 3], 0.f);
  ((float4*)out)[i] = v;
}

// ---------------------------------------------------------------------------
extern "C" void kernel_launch(void* const* d_in, const int* in_sizes, int n_in,
                              void* d_out, int out_size, void* d_ws, size_t ws_size,
                              hipStream_t stream) {
  const float* x    = (const float*)d_in[0];
  const int*   rows = (const int*)d_in[1];
  const int*   cols = (const int*)d_in[2];
  const float* vals = (const float*)d_in[3];
  const float* W    = (const float*)d_in[4];
  const float* b    = (const float*)d_in[5];
  float* out     = (float*)d_out;
  float* support = (float*)d_ws;          // N_NODES * OUT_DIM f32 = 51.2 MB
  const int n_edges = in_sizes[1];
  const int n4 = out_size / 4;

  gcn_zero<<<(n4 + 255) / 256, 256, 0, stream>>>(out, n4);
  gcn_gemm_wmma<<<(GCN_ROW_TILES + 7) / 8, 256, 0, stream>>>(x, W, support);
  gcn_edge_spmm<<<4096, 256, 0, stream>>>(support, rows, cols, vals, out, n_edges);
  gcn_bias_relu<<<(n4 + 255) / 256, 256, 0, stream>>>(out, b, n4);
}
